// FermiNet_spin_283467842325
// MI455X (gfx1250) — matrix operand order repaired and checked by reading, not compiled, not run on hardware
//
#include <hip/hip_runtime.h>
#include <math.h>

// ---------------------------------------------------------------------------
// FermiNet-style network on gfx1250 (MI455X).
//
// Roofline reasoning: the pair stream is (1024*1024) rows x 3 layers of
// 32x32 matmul+softplus ~= 6.4 GFLOP matmul + ~100M exp/log. Materializing
// tp in HBM would cost ~1 GB of traffic; instead the whole tp recurrence is
// kept in WMMA C/D register fragments per 16-pair tile (only the row/col
// SUM statistics, ~1 MB, ever hit memory). Matmuls use
// v_wmma_f32_16x16x32_f16 (M=16 pairs, N=16+16 channels, K=32).
//
// The tp kernel is VALU/TRANS bound (96 softplus per j vs 6 WMMAs), so
// softplus uses raw v_exp_f32/v_log_f32 (base-2 hw transcendentals) with no
// libm denormal fixups: the log argument 1+2^(-|x|*log2e) is in [1,2].
// ---------------------------------------------------------------------------

typedef _Float16 v16h __attribute__((ext_vector_type(16)));
typedef _Float16 v8h  __attribute__((ext_vector_type(8)));
typedef float    v8f  __attribute__((ext_vector_type(8)));

#define NPART 1024
#define PI2_OVER_L 0.62831853071795864769f   // 2*pi/10
#define LOG2E 1.4426950408889634f
#define LN2   0.6931471805599453f

// workspace layout in floats
#define OFF_CA   0                 // 1024*3
#define OFF_SA   3072              // 1024*3
#define OFF_RS0  6144              // 1024*8   rowsum of tpstream0 (stage 0)
#define OFF_CS0  14336             // 1024*8   colsum of tpstream0
#define OFF_RSS  22528             // 3*1024*32 rowsums stages 1..3
#define OFF_CSS  120832            // 3*1024*32 colsums stages 1..3
#define OFF_SP   219136            // 1024*64  sp stream accumulator
#define OFF_UP   284672            // 64
#define OFF_DN   284736            // 64
#define WS_END   284800

// ---------------------------------------------------------------------------
// helpers
// ---------------------------------------------------------------------------
__device__ __forceinline__ v8f wmma16x16x32(v16h a, v16h b, v8f c) {
  return __builtin_amdgcn_wmma_f32_16x16x32_f16(
      /*neg_a=*/false, a, /*neg_b=*/false, b,
      /*c_mod=*/(short)0, c, /*reuse_a=*/false, /*reuse_b=*/false);
}

__device__ __forceinline__ v8f splat8(float x) {
  v8f r;
#pragma unroll
  for (int v = 0; v < 8; ++v) r[v] = x;
  return r;
}

// softplus via raw hw transcendentals (v_exp_f32 = 2^x, v_log_f32 = log2):
//   softplus(x) = max(x,0) + ln2 * log2(1 + 2^(-|x|*log2e))
// argument of log2 is in [1,2] -> no denormal fixup needed.
__device__ __forceinline__ float softplus1(float x) {
  float e = __builtin_amdgcn_exp2f(fabsf(x) * (-LOG2E));
  float l = __builtin_amdgcn_logf(1.0f + e);
  return fmaxf(x, 0.f) + LN2 * l;
}

__device__ __forceinline__ v8f softplus8(v8f x) {
  v8f r;
#pragma unroll
  for (int v = 0; v < 8; ++v) r[v] = softplus1(x[v]);
  return r;
}

// A fragment (16x32 f16) load from an LDS row pointer.
// Lane L: M = L&15, h = L>>4. Element s holds K = h*16 + s (contiguous,
// matching the B fragments)  -> two 16B LDS loads.
__device__ __forceinline__ v16h load_a_frag(const _Float16* row, int h) {
  v8h c0 = *(const v8h*)(row + h * 16);
  v8h c1 = *(const v8h*)(row + h * 16 + 8);
  v16h a;
#pragma unroll
  for (int t = 0; t < 8; ++t) { a[t] = c0[t]; a[8 + t] = c1[t]; }
  return a;
}

// B fragment (32x16 f16) from a row-major f32 weight matrix W[k][ncols],
// output column `col`, zero-padded above kvalid.
__device__ __forceinline__ v16h load_b_frag(const float* W, int ncols,
                                            int col, int h, int kvalid) {
  v16h b;
#pragma unroll
  for (int t = 0; t < 16; ++t) {
    int k = h * 16 + t;
    float w = (k < kvalid) ? W[k * ncols + col] : 0.f;
    b[t] = (_Float16)w;
  }
  return b;
}

// wave-local LDS store->load fence (cross-lane within a wave)
__device__ __forceinline__ void lds_fence_wave() {
  __builtin_amdgcn_wave_barrier();
  asm volatile("s_wait_dscnt 0" ::: "memory");
  __builtin_amdgcn_wave_barrier();
}

// store a D-layout (16x32) f32 tile pair into an LDS tile as f16
__device__ __forceinline__ void store_tile(_Float16* base /*[16][40]*/,
                                           v8f lo, v8f hi, int n, int h) {
#pragma unroll
  for (int v = 0; v < 8; ++v) {
    int r = v + 8 * h;
    base[r * 40 + n]      = (_Float16)lo[v];
    base[r * 40 + 16 + n] = (_Float16)hi[v];
  }
}

// column-sum (over the 16 M rows) of a D-fragment pair -> 32 atomics at dst
__device__ __forceinline__ void colsum_atomic(float* dst, v8f lo, v8f hi,
                                              int lane) {
  float t = 0.f, u = 0.f;
#pragma unroll
  for (int v = 0; v < 8; ++v) { t += lo[v]; u += hi[v]; }
  t += __shfl_xor(t, 16, 32);
  u += __shfl_xor(u, 16, 32);
  int n = lane & 15;
  if (lane < 16) atomicAdd(dst + n, t);
  else           atomicAdd(dst + 16 + n, u);
}

// flush row accumulators (D layout) to rowsum buffer with atomics
__device__ __forceinline__ void flush_rows(float* dst, v8f lo, v8f hi,
                                           int i0, int n, int h) {
#pragma unroll
  for (int v = 0; v < 8; ++v) {
    int i = i0 + v + 8 * h;
    atomicAdd(dst + i * 32 + n, lo[v]);
    atomicAdd(dst + i * 32 + 16 + n, hi[v]);
  }
}

// ---------------------------------------------------------------------------
// kernel 0: per-particle trig tables  ca/sa = cos/sin(2*pi*x/L)
// ---------------------------------------------------------------------------
__global__ void trig_kernel(const float* __restrict__ x,
                            float* __restrict__ ca, float* __restrict__ sa) {
  int idx = blockIdx.x * blockDim.x + threadIdx.x;
  if (idx < NPART * 3) {
    float a = x[idx] * PI2_OVER_L;
    ca[idx] = __cosf(a);
    sa[idx] = __sinf(a);
  }
}

// ---------------------------------------------------------------------------
// kernel 1: full tp pipeline over pair tiles, in registers.
// grid (64 i-tiles, 8 j-groups), 128 threads = 4 waves; each wave owns
// 16 i-rows x 32 j-columns, pipelining 3 WMMA layers per j.
// ---------------------------------------------------------------------------
__global__ __launch_bounds__(128) void tp_stream_kernel(
    const float* __restrict__ ca, const float* __restrict__ sa,
    const float* __restrict__ tw0, const float* __restrict__ tb0,
    const float* __restrict__ tw12, const float* __restrict__ tb12,
    float* __restrict__ rs0, float* __restrict__ cs0,
    float* __restrict__ rss, float* __restrict__ css) {
  __shared__ _Float16 tF[4][16][40];  // stage-0 features (K padded 8->32)
  __shared__ _Float16 tT[4][16][40];  // tp restage between layers

  const int lane = threadIdx.x & 31;
  const int wv   = threadIdx.x >> 5;
  const int n    = lane & 15;
  const int h    = lane >> 4;
  const int i0   = blockIdx.x * 16;
  const int jbase = blockIdx.y * 128 + wv * 32;

  // zero the K=8..31 padding of the feature tile once
  for (int idx = threadIdx.x; idx < 4 * 16; idx += 128)
#pragma unroll
    for (int k = 8; k < 40; ++k) tF[idx >> 4][idx & 15][k] = (_Float16)0.f;
  __syncthreads();

  // weight fragments (held in VGPRs for the whole kernel)
  v16h B0l = load_b_frag(tw0,          32, n,      h, 8);
  v16h B0h = load_b_frag(tw0,          32, 16 + n, h, 8);
  v16h B1l = load_b_frag(tw12,         32, n,      h, 32);
  v16h B1h = load_b_frag(tw12,         32, 16 + n, h, 32);
  v16h B2l = load_b_frag(tw12 + 1024,  32, n,      h, 32);
  v16h B2h = load_b_frag(tw12 + 1024,  32, 16 + n, h, 32);
  const float b0l = tb0[n],        b0h = tb0[16 + n];
  const float b1l = tb12[n],       b1h = tb12[16 + n];
  const float b2l = tb12[32 + n],  b2h = tb12[48 + n];

  // row-sum accumulators per stage (D layout)
  v8f r1l = splat8(0.f), r1h = splat8(0.f);
  v8f r2l = splat8(0.f), r2h = splat8(0.f);
  v8f r3l = splat8(0.f), r3h = splat8(0.f);
  float rsa[8];
#pragma unroll
  for (int t = 0; t < 8; ++t) rsa[t] = 0.f;

  // per-particle trig for this lane's row i = i0 + n
  const int ii = i0 + n;
  float cai0 = ca[ii * 3 + 0], cai1 = ca[ii * 3 + 1], cai2 = ca[ii * 3 + 2];
  float sai0 = sa[ii * 3 + 0], sai1 = sa[ii * 3 + 1], sai2 = sa[ii * 3 + 2];

  for (int jj = 0; jj < 32; ++jj) {
    const int j = jbase + jj;
    // ---- stage 0: pairwise features via angle subtraction ----
    float caj0 = ca[j * 3 + 0], caj1 = ca[j * 3 + 1], caj2 = ca[j * 3 + 2];
    float saj0 = sa[j * 3 + 0], saj1 = sa[j * 3 + 1], saj2 = sa[j * 3 + 2];
    float c0 = cai0 * caj0 + sai0 * saj0;
    float c1 = cai1 * caj1 + sai1 * saj1;
    float c2 = cai2 * caj2 + sai2 * saj2;
    float s0 = sai0 * caj0 - cai0 * saj0;
    float s1 = sai1 * caj1 - cai1 * saj1;
    float s2 = sai2 * caj2 - cai2 * saj2;
    float ds = sqrtf(s0 * s0 + s1 * s1 + s2 * s2);
    float dc = sqrtf(c0 * c0 + c1 * c1 + c2 * c2);
    if (ii == j) { ds = 0.f; dc = 0.f; }  // diagonal: c=1, s=0 automatically
    float f[8] = {c0, c1, c2, s0, s1, s2, ds, dc};

    if (lane < 16) {
#pragma unroll
      for (int t = 0; t < 8; ++t) tF[wv][lane][t] = (_Float16)f[t];
    }
#pragma unroll
    for (int t = 0; t < 8; ++t) rsa[t] += f[t];  // only lanes<16 used later
    lds_fence_wave();

    // stage-0 column sums (lanes 0..15 and 16..31 hold identical copies)
#pragma unroll
    for (int t = 0; t < 8; ++t) {
      float v = f[t];
      v += __shfl_xor(v, 8, 32);
      v += __shfl_xor(v, 4, 32);
      v += __shfl_xor(v, 2, 32);
      v += __shfl_xor(v, 1, 32);
      f[t] = v;
    }
    if (lane == 0) {
#pragma unroll
      for (int t = 0; t < 8; ++t) atomicAdd(&cs0[j * 8 + t], f[t]);
    }

    // ---- layer 0 (K=8 padded to 32): tp = softplus(feat @ W0 + b0) ----
    v16h A = load_a_frag(&tF[wv][n][0], h);
    v8f dl = wmma16x16x32(A, B0l, splat8(b0l));
    v8f dh = wmma16x16x32(A, B0h, splat8(b0h));
    v8f tl = softplus8(dl);
    v8f th = softplus8(dh);
    r1l += tl; r1h += th;
    colsum_atomic(&css[0 * 32768 + j * 32], tl, th, lane);
    store_tile(&tT[wv][0][0], tl, th, n, h);
    lds_fence_wave();

    // ---- layer 1: tp += softplus(tp @ W1 + b1) ----
    A  = load_a_frag(&tT[wv][n][0], h);
    dl = wmma16x16x32(A, B1l, splat8(b1l));
    dh = wmma16x16x32(A, B1h, splat8(b1h));
    tl += softplus8(dl);
    th += softplus8(dh);
    r2l += tl; r2h += th;
    colsum_atomic(&css[1 * 32768 + j * 32], tl, th, lane);
    store_tile(&tT[wv][0][0], tl, th, n, h);
    lds_fence_wave();

    // ---- layer 2: tp += softplus(tp @ W2 + b2) ----
    A  = load_a_frag(&tT[wv][n][0], h);
    dl = wmma16x16x32(A, B2l, splat8(b2l));
    dh = wmma16x16x32(A, B2h, splat8(b2h));
    tl += softplus8(dl);
    th += softplus8(dh);
    r3l += tl; r3h += th;
    colsum_atomic(&css[2 * 32768 + j * 32], tl, th, lane);
  }

  // flush row sums
  if (lane < 16) {
#pragma unroll
    for (int t = 0; t < 8; ++t) atomicAdd(&rs0[(i0 + lane) * 8 + t], rsa[t]);
  }
  flush_rows(rss + 0 * 32768, r1l, r1h, i0, n, h);
  flush_rows(rss + 1 * 32768, r2l, r2h, i0, n, h);
  flush_rows(rss + 2 * 32768, r3l, r3h, i0, n, h);
}

// ---------------------------------------------------------------------------
// kernel 2: spin-channel means of sp (64 threads, 1 block)
// ---------------------------------------------------------------------------
__global__ void sp_mean_kernel(const float* __restrict__ sp,
                               float* __restrict__ up, float* __restrict__ dn) {
  int t = threadIdx.x;  // 0..63
  float su = 0.f, sd = 0.f;
  for (int i = 0; i < 512; ++i) {
    su += sp[i * 64 + t];
    sd += sp[(512 + i) * 64 + t];
  }
  up[t] = su * (1.f / 512.f);
  dn[t] = sd * (1.f / 512.f);
}

// ---------------------------------------------------------------------------
// kernel 3: one sp layer: sp (+)= softplus(feat @ W + b)
// grid 64 (16-row tiles), 128 threads = 4 waves; wave w -> cols [16w,16w+16)
// ---------------------------------------------------------------------------
__global__ __launch_bounds__(128) void sp_layer_kernel(
    int layer, float* __restrict__ sp,
    const float* __restrict__ up, const float* __restrict__ dn,
    const float* __restrict__ rs0, const float* __restrict__ cs0,
    const float* __restrict__ rss, const float* __restrict__ css,
    const float* __restrict__ W, const float* __restrict__ B) {
  __shared__ _Float16 fs[16][264];  // 16 rows x 256 (padded) features, f16
  const int lane = threadIdx.x & 31;
  const int wv   = threadIdx.x >> 5;
  const int n    = lane & 15;
  const int h    = lane >> 4;
  const int i0   = blockIdx.x * 16;
  const float invN = 1.f / 1024.f;

  if (layer == 0) {
    // f = [sp(3)=0, up(3)=0, dn(3)=0, tpmean_j(8), tpmean_i(8), pad] -> 32
    for (int idx = threadIdx.x; idx < 16 * 32; idx += 128) {
      int r = idx >> 5, k = idx & 31;
      float v = 0.f;
      if (k >= 9 && k < 17)       v = rs0[(i0 + r) * 8 + (k - 9)] * invN;
      else if (k >= 17 && k < 25) v = cs0[(i0 + r) * 8 + (k - 17)] * invN;
      fs[r][k] = (_Float16)v;
    }
  } else {
    const float* rs = rss + (size_t)(layer - 1) * 32768;
    const float* cs = css + (size_t)(layer - 1) * 32768;
    for (int idx = threadIdx.x; idx < 16 * 256; idx += 128) {
      int r = idx >> 8, k = idx & 255;
      float v;
      if (k < 64)       v = sp[(i0 + r) * 64 + k];
      else if (k < 128) v = up[k - 64];
      else if (k < 192) v = dn[k - 128];
      else if (k < 224) v = rs[(i0 + r) * 32 + (k - 192)] * invN;
      else              v = cs[(i0 + r) * 32 + (k - 224)] * invN;
      fs[r][k] = (_Float16)v;
    }
  }
  __syncthreads();

  const int col = wv * 16 + n;
  v8f acc = splat8(B[col]);
  const int nchunks = (layer == 0) ? 1 : 8;
  const int kvalid  = (layer == 0) ? 25 : 256;
  for (int kc = 0; kc < nchunks; ++kc) {
    v16h a = load_a_frag(&fs[n][kc * 32], h);
    v16h b;
#pragma unroll
    for (int t = 0; t < 16; ++t) {
      int k = kc * 32 + h * 16 + t;
      float w = (k < kvalid) ? W[k * 64 + col] : 0.f;
      b[t] = (_Float16)w;
    }
    acc = wmma16x16x32(a, b, acc);
  }
#pragma unroll
  for (int v = 0; v < 8; ++v) {
    int i = i0 + v + 8 * h;
    float val = softplus1(acc[v]);
    if (layer == 0) sp[i * 64 + col] = val;
    else            sp[i * 64 + col] += val;
  }
}

// ---------------------------------------------------------------------------
// kernel 4: out = x + sp @ fin_w + fin_b
// ---------------------------------------------------------------------------
__global__ void out_kernel(const float* __restrict__ x,
                           const float* __restrict__ sp,
                           const float* __restrict__ fw,
                           const float* __restrict__ fb,
                           float* __restrict__ out) {
  int i = blockIdx.x * blockDim.x + threadIdx.x;
  if (i < NPART) {
    float a0 = fb[0], a1 = fb[1], a2 = fb[2];
#pragma unroll 4
    for (int k = 0; k < 64; ++k) {
      float s = sp[i * 64 + k];
      a0 = fmaf(s, fw[k * 3 + 0], a0);
      a1 = fmaf(s, fw[k * 3 + 1], a1);
      a2 = fmaf(s, fw[k * 3 + 2], a2);
    }
    out[i * 3 + 0] = x[i * 3 + 0] + a0;
    out[i * 3 + 1] = x[i * 3 + 1] + a1;
    out[i * 3 + 2] = x[i * 3 + 2] + a2;
  }
}

// ---------------------------------------------------------------------------
extern "C" void kernel_launch(void* const* d_in, const int* in_sizes, int n_in,
                              void* d_out, int out_size, void* d_ws,
                              size_t ws_size, hipStream_t stream) {
  (void)in_sizes; (void)n_in; (void)out_size; (void)ws_size;
  const float* x     = (const float*)d_in[0];
  const float* sp_w0 = (const float*)d_in[1];
  const float* sp_b0 = (const float*)d_in[2];
  const float* sp_w  = (const float*)d_in[3];
  const float* sp_b  = (const float*)d_in[4];
  const float* tp_w0 = (const float*)d_in[5];
  const float* tp_b0 = (const float*)d_in[6];
  const float* tp_w  = (const float*)d_in[7];
  const float* tp_b  = (const float*)d_in[8];
  const float* fin_w = (const float*)d_in[9];
  const float* fin_b = (const float*)d_in[10];

  float* ws  = (float*)d_ws;
  float* ca  = ws + OFF_CA;
  float* sa  = ws + OFF_SA;
  float* rs0 = ws + OFF_RS0;
  float* cs0 = ws + OFF_CS0;
  float* rss = ws + OFF_RSS;
  float* css = ws + OFF_CSS;
  float* sp  = ws + OFF_SP;
  float* up  = ws + OFF_UP;
  float* dn  = ws + OFF_DN;

  // zero all accumulator buffers + sp (deterministic, capture-safe)
  hipMemsetAsync(rs0, 0, (size_t)(WS_END - OFF_RS0) * sizeof(float), stream);

  trig_kernel<<<12, 256, 0, stream>>>(x, ca, sa);
  tp_stream_kernel<<<dim3(64, 8), 128, 0, stream>>>(
      ca, sa, tp_w0, tp_b0, tp_w, tp_b, rs0, cs0, rss, css);

  for (int l = 0; l < 4; ++l) {
    sp_mean_kernel<<<1, 64, 0, stream>>>(sp, up, dn);
    const float* W = (l == 0) ? sp_w0 : sp_w + (size_t)(l - 1) * 256 * 64;
    const float* B = (l == 0) ? sp_b0 : sp_b + (size_t)(l - 1) * 64;
    sp_layer_kernel<<<64, 128, 0, stream>>>(l, sp, up, dn, rs0, cs0, rss, css,
                                            W, B);
  }

  out_kernel<<<4, 256, 0, stream>>>(x, sp, fin_w, fin_b, (float*)d_out);
}
